// LigerFusedLinearDPOLoss_15204184227923
// MI455X (gfx1250) — compile-verified, matrix operand order checked by reading, not run on hardware
//
#include <hip/hip_runtime.h>
#include <hip/hip_bf16.h>

// Problem constants (from the reference).
#define H_DIM  4096
#define V_DIM  32000
#define BT     4096     // B*T tokens
#define SEQ_T  512
#define MBLK   256      // token rows per workgroup (GEMM)
#define NPANEL 256      // vocab columns per workgroup panel
#define NPANELS (V_DIM / NPANEL)   // 125
#define BK     64       // K tile (two 16x16x32 WMMA K-steps)
#define IGNORE_INDEX (-100)
#define BETA   0.1f

typedef __attribute__((ext_vector_type(16))) __bf16 bf16x16;
typedef __attribute__((ext_vector_type(8)))  float  f32x8;
typedef __attribute__((ext_vector_type(4)))  int    i32x4;

// ---------------------------------------------------------------------------
// fp32 -> packed bf16x2 (hardware cvt_pk if the builtin exists)
// ---------------------------------------------------------------------------
#if __has_builtin(__builtin_amdgcn_cvt_pk_bf16_f32)
typedef __attribute__((ext_vector_type(2))) __bf16 bf16x2;
__device__ __forceinline__ unsigned int pack_bf16x2(float lo, float hi) {
    union { bf16x2 v; unsigned int u; } c;
    c.v = __builtin_amdgcn_cvt_pk_bf16_f32(lo, hi);
    return c.u;
}
#else
__device__ __forceinline__ unsigned short f32_to_bf16(float f) {
    unsigned int u = __float_as_uint(f);
    u += 0x7fffu + ((u >> 16) & 1u);          // round-to-nearest-even
    return (unsigned short)(u >> 16);
}
__device__ __forceinline__ unsigned int pack_bf16x2(float lo, float hi) {
    return (unsigned int)f32_to_bf16(lo) | ((unsigned int)f32_to_bf16(hi) << 16);
}
#endif

__device__ __forceinline__ float bf_lo(unsigned int u) { return __uint_as_float(u << 16); }
__device__ __forceinline__ float bf_hi(unsigned int u) { return __uint_as_float(u & 0xffff0000u); }

// ---------------------------------------------------------------------------
// 16-byte global -> LDS copy: async-to-LDS (ASYNCcnt) when available.
// Builtin signature (probe-verified): (v4i AS1*, v4i AS3*, imm, imm).
// ---------------------------------------------------------------------------
#if __has_builtin(__builtin_amdgcn_global_load_async_to_lds_b128) && \
    __has_builtin(__builtin_amdgcn_s_wait_asynccnt)
#define USE_ASYNC_LDS 1
#else
#define USE_ASYNC_LDS 0
#endif

__device__ __forceinline__ void copy16_g2l(const unsigned short* g, unsigned short* l) {
#if USE_ASYNC_LDS
    const __attribute__((address_space(1))) void* g1v =
        (const __attribute__((address_space(1))) void*)g;
    __attribute__((address_space(3))) void* l3v =
        (__attribute__((address_space(3))) void*)l;
    __builtin_amdgcn_global_load_async_to_lds_b128(
        (__attribute__((address_space(1))) i32x4*)g1v,
        (__attribute__((address_space(3))) i32x4*)l3v, 0, 0);
#else
    *(uint4*)l = *(const uint4*)g;
#endif
}

// ---------------------------------------------------------------------------
// Kernel 0: elementwise fp32 -> bf16 (one float4 -> uint2 per thread)
// ---------------------------------------------------------------------------
__global__ __launch_bounds__(256)
void f32_to_bf16_kernel(const float* __restrict__ src, unsigned int* __restrict__ dst, long n4) {
    const long i = (long)blockIdx.x * 256 + threadIdx.x;
    if (i < n4) {
        const float4 v = ((const float4*)src)[i];
        uint2 o;
        o.x = pack_bf16x2(v.x, v.y);
        o.y = pack_bf16x2(v.z, v.w);
        ((uint2*)dst)[i] = o;
    }
}

// ---------------------------------------------------------------------------
// Kernel 1: bf16 WMMA GEMM (256x256 tile, double-buffered async LDS staging)
//           + per-panel (max, sumexp) partials.
// grid: (BT/256, NPANELS, 2 models), block 512 (16 wave32s).
// Wave (rt = w&7, ch = w>>3): two row tiles {2rt, 2rt+1}, column half ch
// (8 x 16-col tiles). LDS tiles stored as XOR-swizzled 16B chunks
// (8 chunks/row, stride 128B): conflict-free b128 copies and fragment reads.
// Double buffering: stage k+1 via async-to-LDS while computing tile k;
// s_wait_asynccnt(8) retires only the 8 oldest (current-tile) copies.
// ---------------------------------------------------------------------------
__global__ __launch_bounds__(512)
void dpo_gemm_lse_kernel(const unsigned short* __restrict__ xbf,  // [2][BT][H]
                         const unsigned short* __restrict__ wbf,  // [2][V][H]
                         const float* __restrict__ b0,
                         const float* __restrict__ b1,
                         float* __restrict__ partials)            // [2][BT][NPANELS][2]
{
    const int model = blockIdx.z;
    const unsigned short* __restrict__ xb = xbf + (size_t)model * BT * H_DIM;
    const unsigned short* __restrict__ wb = wbf + (size_t)model * V_DIM * H_DIM;
    const float* __restrict__ bias = model ? b1 : b0;

    const int row0 = blockIdx.x * MBLK;
    const int n0   = blockIdx.y * NPANEL;

    __shared__ __align__(16) unsigned short lA[2][MBLK * BK];    // 2 x 32 KB
    __shared__ __align__(16) unsigned short lB[2][NPANEL * BK];  // 2 x 32 KB
    __shared__ float redM[16][16][2];
    __shared__ float redS[16][16][2];

    const int t    = threadIdx.x;
    const int lane = t & 31;
    const int wave = t >> 5;
    const int rt   = wave & 7;   // row-tile pair index 0..7 (tiles 2rt, 2rt+1)
    const int ch   = wave >> 3;  // column half 0..1
    const int frow  = lane & 15; // M (A) or N (B) index within tile
    const int fhalf = lane >> 4; // K-half per ISA 16-bit layout

    f32x8 acc[2][8];
    #pragma unroll
    for (int rr = 0; rr < 2; ++rr)
        #pragma unroll
        for (int ct = 0; ct < 8; ++ct)
            acc[rr][ct] = (f32x8){0.f,0.f,0.f,0.f,0.f,0.f,0.f,0.f};

    // Stage one K-tile (A: 256x8 chunks, B: 256x8 chunks -> 4+4 b128/thread).
    auto stage = [&](int buf, int k0) {
        #pragma unroll
        for (int i = 0; i < 4; ++i) {
            const int id  = i * 512 + t;
            const int row = id >> 3;
            const int c   = id & 7;
            copy16_g2l(&xb[(size_t)(row0 + row) * H_DIM + k0 + c * 8],
                       &lA[buf][row * BK + ((c ^ (row & 7)) * 8)]);
        }
        #pragma unroll
        for (int i = 0; i < 4; ++i) {
            const int id  = i * 512 + t;
            const int row = id >> 3;
            const int c   = id & 7;
            copy16_g2l(&wb[(size_t)(n0 + row) * H_DIM + k0 + c * 8],
                       &lB[buf][row * BK + ((c ^ (row & 7)) * 8)]);
        }
    };

    stage(0, 0);  // prologue

    int it = 0;
    for (int k0 = 0; k0 < H_DIM; k0 += BK, ++it) {
        const int cur = it & 1;
        const unsigned short* __restrict__ lAc = lA[cur];
        const unsigned short* __restrict__ lBc = lB[cur];
        const bool have_next = (k0 + BK) < H_DIM;
        if (have_next) stage(cur ^ 1, k0 + BK);
#if USE_ASYNC_LDS
        if (have_next) __builtin_amdgcn_s_wait_asynccnt(8);  // oldest 8 = current tile
        else           __builtin_amdgcn_s_wait_asynccnt(0);
#endif
        __syncthreads();

        #pragma unroll
        for (int s = 0; s < 2; ++s) {
            union Frag { unsigned int u[8]; bf16x16 v; };
            const int c1 = s * 4 + fhalf;  // first 16B chunk of this fragment
            Frag fa[2];
            #pragma unroll
            for (int rr = 0; rr < 2; ++rr) {
                const int arow = (rt * 2 + rr) * 16 + frow;
                const unsigned short* ab = &lAc[arow * BK];
                const int sw = arow & 7;
                *(uint4*)&fa[rr].u[0] = *(const uint4*)&ab[((c1)     ^ sw) * 8];
                *(uint4*)&fa[rr].u[4] = *(const uint4*)&ab[((c1 + 2) ^ sw) * 8];
            }
            #pragma unroll
            for (int ct = 0; ct < 8; ++ct) {
                Frag fb;
                const int brow = ch * 128 + ct * 16 + frow;
                const unsigned short* bb = &lBc[brow * BK];
                const int sw = brow & 7;
                *(uint4*)&fb.u[0] = *(const uint4*)&bb[((c1)     ^ sw) * 8];
                *(uint4*)&fb.u[4] = *(const uint4*)&bb[((c1 + 2) ^ sw) * 8];
                acc[0][ct] = __builtin_amdgcn_wmma_f32_16x16x32_bf16(
                                 false, fa[0].v, false, fb.v, (short)0, acc[0][ct], false, false);
                acc[1][ct] = __builtin_amdgcn_wmma_f32_16x16x32_bf16(
                                 false, fa[1].v, false, fb.v, (short)0, acc[1][ct], false, false);
            }
        }
        __syncthreads();  // protects buffer `cur` before it is restaged next iter
    }

    // ---- epilogue: bias add, register-resident log-sum-exp over the panel ----
    #pragma unroll
    for (int ct = 0; ct < 8; ++ct) {
        const float bv = bias[n0 + ch * 128 + ct * 16 + frow];
        #pragma unroll
        for (int rr = 0; rr < 2; ++rr)
            #pragma unroll
            for (int g = 0; g < 8; ++g) acc[rr][ct][g] += bv;
    }

    // per-row max over this wave's 128 columns (xor-shuffle in 16-lane half)
    #pragma unroll
    for (int rr = 0; rr < 2; ++rr) {
        #pragma unroll
        for (int g = 0; g < 8; ++g) {
            float mm = acc[rr][0][g];
            #pragma unroll
            for (int ct = 1; ct < 8; ++ct) mm = fmaxf(mm, acc[rr][ct][g]);
            #pragma unroll
            for (int off = 1; off < 16; off <<= 1) mm = fmaxf(mm, __shfl_xor(mm, off, 32));
            if (frow == 0) redM[rt * 2 + rr][g + 8 * fhalf][ch] = mm;
        }
    }
    __syncthreads();

    // full row max, then per-row sumexp partial over this wave's 128 columns
    #pragma unroll
    for (int rr = 0; rr < 2; ++rr) {
        #pragma unroll
        for (int g = 0; g < 8; ++g) {
            const int r = g + 8 * fhalf;
            const float M = fmaxf(redM[rt * 2 + rr][r][0], redM[rt * 2 + rr][r][1]);
            float s = 0.f;
            #pragma unroll
            for (int ct = 0; ct < 8; ++ct) s += __expf(acc[rr][ct][g] - M);
            #pragma unroll
            for (int off = 1; off < 16; off <<= 1) s += __shfl_xor(s, off, 32);
            if (frow == 0) redS[rt * 2 + rr][r][ch] = s;
        }
    }
    __syncthreads();

    if (t < MBLK) {
        const int rt2 = t >> 4, r = t & 15;
        const float mm = fmaxf(redM[rt2][r][0], redM[rt2][r][1]);
        const float ss = redS[rt2][r][0] + redS[rt2][r][1];
        const size_t pidx = (((size_t)model * BT + row0 + t) * NPANELS + blockIdx.y) * 2;
        partials[pidx]     = mm;
        partials[pidx + 1] = ss;
    }
}

// ---------------------------------------------------------------------------
// Kernel 2a: target logits, one wave per (model, token) from the SAME bf16 data
// grid: 2*BT/8 blocks, 256 threads (8 waves)
// ---------------------------------------------------------------------------
__global__ __launch_bounds__(256)
void tgt_logit_kernel(const unsigned short* __restrict__ xbf,
                      const unsigned short* __restrict__ wbf,
                      const float* __restrict__ b0, const float* __restrict__ b1,
                      const int* __restrict__ target,
                      float* __restrict__ tgt_logit)   // [2][BT]
{
    const int wave = threadIdx.x >> 5, lane = threadIdx.x & 31;
    const int gidx = blockIdx.x * 8 + wave;    // 0 .. 2*BT-1
    const int model = gidx >= BT;
    const int tok   = gidx & (BT - 1);
    const int tg    = target[tok];

    float r = 0.f;
    if (tg != IGNORE_INDEX) {
        const unsigned short* xr = xbf + ((size_t)model * BT + tok) * H_DIM;
        const unsigned short* wr = wbf + ((size_t)model * V_DIM + tg) * H_DIM;
        float acc = 0.f;
        for (int k = lane * 8; k < H_DIM; k += 32 * 8) {
            const uint4 xa = *(const uint4*)&xr[k];
            const uint4 wa = *(const uint4*)&wr[k];
            acc += bf_lo(xa.x) * bf_lo(wa.x) + bf_hi(xa.x) * bf_hi(wa.x);
            acc += bf_lo(xa.y) * bf_lo(wa.y) + bf_hi(xa.y) * bf_hi(wa.y);
            acc += bf_lo(xa.z) * bf_lo(wa.z) + bf_hi(xa.z) * bf_hi(wa.z);
            acc += bf_lo(xa.w) * bf_lo(wa.w) + bf_hi(xa.w) * bf_hi(wa.w);
        }
        #pragma unroll
        for (int off = 1; off < 32; off <<= 1) acc += __shfl_xor(acc, off, 32);
        const float* bias = model ? b1 : b0;
        r = acc + bias[tg];
    }
    if (lane == 0) tgt_logit[gidx] = r;
}

// ---------------------------------------------------------------------------
// Kernel 2b: combine panel LSE partials -> per-sequence sum of target log-probs
// ---------------------------------------------------------------------------
__global__ __launch_bounds__(256)
void dpo_seq_reduce_kernel(const float* __restrict__ partials,
                           const float* __restrict__ tgt_logit,
                           const int*   __restrict__ target,
                           float* __restrict__ logps) // [2][8]
{
    const int model = blockIdx.x >> 3;
    const int seq   = blockIdx.x & 7;
    const int t     = threadIdx.x;

    float acc = 0.f;
    for (int i = t; i < SEQ_T; i += 256) {
        const int tok = seq * SEQ_T + i;
        const float* p = &partials[(((size_t)model * BT + tok) * NPANELS) * 2];
        float M = -3.402823466e38f, S = 0.f;
        for (int pa = 0; pa < NPANELS; ++pa) {
            const float m = p[pa * 2], s = p[pa * 2 + 1];
            if (m > M) { S = S * __expf(M - m) + s; M = m; }
            else       { S += s * __expf(m - M); }
        }
        const float logZ = M + __logf(S);
        const int tg = target[tok];
        if (tg != IGNORE_INDEX)
            acc += tgt_logit[(size_t)model * BT + tok] - logZ;
    }
    __shared__ float red[256];
    red[t] = acc;
    __syncthreads();
    for (int o = 128; o > 0; o >>= 1) {
        if (t < o) red[t] += red[t + o];
        __syncthreads();
    }
    if (t == 0) logps[blockIdx.x] = red[0];
}

// ---------------------------------------------------------------------------
// Kernel 3: DPO loss over 4 (chosen, rejected) pairs
// ---------------------------------------------------------------------------
__device__ __forceinline__ float log_sigmoid(float v) {
    return (v >= 0.f) ? -log1pf(__expf(-v)) : (v - log1pf(__expf(v)));
}
__global__ void dpo_loss_kernel(const float* __restrict__ logps, float* __restrict__ out) {
    if (threadIdx.x == 0) {
        float loss = 0.f;
        #pragma unroll
        for (int i = 0; i < 4; ++i) {
            const float c  = logps[i];       // policy chosen
            const float r  = logps[4 + i];   // policy rejected
            const float rc = logps[8 + i];   // ref chosen
            const float rr = logps[12 + i];  // ref rejected
            const float d  = BETA * ((c - rc) - (r - rr));
            loss += -log_sigmoid(d);
        }
        out[0] = loss * 0.25f;
    }
}

// ---------------------------------------------------------------------------
extern "C" void kernel_launch(void* const* d_in, const int* in_sizes, int n_in,
                              void* d_out, int out_size, void* d_ws, size_t ws_size,
                              hipStream_t stream) {
    const float* x0 = (const float*)d_in[0];  // _input     (BT, H)
    const float* w0 = (const float*)d_in[1];  // weight     (V, H)
    const float* b0 = (const float*)d_in[2];  // bias       (V,)
    const float* x1 = (const float*)d_in[3];  // ref_input
    const float* w1 = (const float*)d_in[4];  // ref_weight
    const float* b1 = (const float*)d_in[5];  // ref_bias
    const int*  tgt = (const int*)d_in[6];    // target     (BT,)

    // workspace layout
    unsigned short* xbf = (unsigned short*)d_ws;                    // [2][BT][H]
    unsigned short* wbf = xbf + (size_t)2 * BT * H_DIM;             // [2][V][H]
    float* partials  = (float*)(wbf + (size_t)2 * V_DIM * H_DIM);   // [2][BT][NPANELS][2]
    float* tgt_logit = partials + (size_t)2 * BT * NPANELS * 2;     // [2][BT]
    float* logps     = tgt_logit + (size_t)2 * BT;                  // [16]

    // 0) one-time fp32 -> bf16 conversion
    const long nx4 = (long)BT * H_DIM / 4;
    const long nw4 = (long)V_DIM * H_DIM / 4;
    f32_to_bf16_kernel<<<(int)((nx4 + 255) / 256), 256, 0, stream>>>(
        x0, (unsigned int*)xbf, nx4);
    f32_to_bf16_kernel<<<(int)((nx4 + 255) / 256), 256, 0, stream>>>(
        x1, (unsigned int*)(xbf + (size_t)BT * H_DIM), nx4);
    f32_to_bf16_kernel<<<(int)((nw4 + 255) / 256), 256, 0, stream>>>(
        w0, (unsigned int*)wbf, nw4);
    f32_to_bf16_kernel<<<(int)((nw4 + 255) / 256), 256, 0, stream>>>(
        w1, (unsigned int*)(wbf + (size_t)V_DIM * H_DIM), nw4);

    // 1) fused GEMM + per-panel LSE partials
    dim3 g1(BT / MBLK, NPANELS, 2);
    dpo_gemm_lse_kernel<<<g1, 512, 0, stream>>>(xbf, wbf, b0, b1, partials);

    // 2) target logits + per-sequence reduction
    tgt_logit_kernel<<<2 * BT / 8, 256, 0, stream>>>(xbf, wbf, b0, b1, tgt, tgt_logit);
    dpo_seq_reduce_kernel<<<16, 256, 0, stream>>>(partials, tgt_logit, tgt, logps);

    // 3) scalar DPO loss
    dpo_loss_kernel<<<1, 32, 0, stream>>>(logps, (float*)d_out);
}
